// S5Block_35381940584731
// MI455X (gfx1250) — matrix operand order repaired and checked by reading, not compiled, not run on hardware
//
#include <hip/hip_runtime.h>

#define BATCH 4
#define SEQL  8192
#define HDIM  1024
#define PDIM  256
#define N1    512                 // 2*PDIM, columns interleaved (2p=re, 2p+1=im)
#define MROWS (BATCH*SEQL)        // 32768
#define LCHUNK 128
#define NCHUNK (SEQL/LCHUNK)      // 64
#define EPSV  1e-6f

static_assert(LCHUNK == 128, "tile row-panel == scan chunk; chunk_scan uses 7 squarings");

typedef __attribute__((ext_vector_type(16))) __bf16 v16bf;
typedef __attribute__((ext_vector_type(8)))  float  v8f;
typedef __attribute__((ext_vector_type(4)))  int    v4i;
typedef __attribute__((address_space(3))) v4i* lds_v4i_ptr;

union BF16Frag { v16bf v; unsigned short s[16]; unsigned int u[8]; };

#ifndef __has_builtin
#define __has_builtin(x) 0
#endif
#if __has_builtin(__builtin_amdgcn_global_load_async_to_lds_b128) && \
    __has_builtin(__builtin_amdgcn_s_wait_asynccnt)
#define HAVE_ASYNC 1
#else
#define HAVE_ASYNC 0
#endif

// copy 16 ushorts (32B) global -> LDS; async path uses CDNA5 ASYNCcnt DMA
__device__ __forceinline__ void cp16b(unsigned short* l, const unsigned short* g) {
#if HAVE_ASYNC
  __builtin_amdgcn_global_load_async_to_lds_b128(
      (v4i*)g, (lds_v4i_ptr)(void*)l, 0, 0);
  __builtin_amdgcn_global_load_async_to_lds_b128(
      (v4i*)(g + 8), (lds_v4i_ptr)(void*)(l + 8), 0, 0);
#else
  *(uint4*)l = *(const uint4*)g;
  *((uint4*)l + 1) = *((const uint4*)g + 1);
#endif
}
__device__ __forceinline__ void cp8b(unsigned short* l, const unsigned short* g) {
#if HAVE_ASYNC
  __builtin_amdgcn_global_load_async_to_lds_b128(
      (v4i*)g, (lds_v4i_ptr)(void*)l, 0, 0);
#else
  *(uint4*)l = *(const uint4*)g;
#endif
}
__device__ __forceinline__ void async_join() {
#if HAVE_ASYNC
  __builtin_amdgcn_s_wait_asynccnt(0);
#endif
}

__device__ __forceinline__ unsigned short f32_to_bf16_rne(float f) {
  unsigned int x = __float_as_uint(f);
  unsigned int r = x + 0x7FFFu + ((x >> 16) & 1u);
  return (unsigned short)(r >> 16);
}
__device__ __forceinline__ unsigned int pack_bf16x2(float lo, float hi) {
  return (unsigned int)f32_to_bf16_rne(lo) | ((unsigned int)f32_to_bf16_rne(hi) << 16);
}
__device__ __forceinline__ float bf16_lo(unsigned int w) {
  return __uint_as_float((w & 0xFFFFu) << 16);
}
__device__ __forceinline__ float bf16_hi(unsigned int w) {
  return __uint_as_float(w & 0xFFFF0000u);
}

// ---------------- prep kernels (weights stored TRANSPOSED, n-major) ----------------

// W1T[n][h], n in [0,512): n=2p -> Re(B_bar)[p][h], n=2p+1 -> Im(B_bar)[p][h]
__global__ void prep_w1t(const float* __restrict__ Lre, const float* __restrict__ Lim,
                         const float* __restrict__ lstep,
                         const float* __restrict__ Bre, const float* __restrict__ Bim,
                         unsigned short* __restrict__ W1T) {
  int idx = blockIdx.x * blockDim.x + threadIdx.x;   // over N1*HDIM
  if (idx >= N1 * HDIM) return;
  int n = idx / HDIM;
  int h = idx % HDIM;
  int p = n >> 1;
  float lr = Lre[p], li = Lim[p];
  float dt = __expf(lstep[p]);
  float e  = __expf(lr * dt);
  float ar = e * __cosf(li * dt);                     // lam_bar
  float ai = e * __sinf(li * dt);
  float nr = ar - 1.0f, ni = ai;                      // lam_bar - 1
  float den = lr * lr + li * li;                      // |lam|^2 (never 0 here)
  float cr = (nr * lr + ni * li) / den;               // (lam_bar-1)/lam
  float ci = (ni * lr - nr * li) / den;
  float br = Bre[p * HDIM + h], bi = Bim[p * HDIM + h];
  float wr = cr * br - ci * bi;                       // Re(B_bar)
  float wi = cr * bi + ci * br;                       // Im(B_bar)
  W1T[(size_t)n * HDIM + h] = f32_to_bf16_rne((n & 1) ? wi : wr);
}

// W2T[h][k], k in [0,512): k=2p -> 2*C_re[h][p], k=2p+1 -> -2*C_im[h][p]
__global__ void prep_w2t(const float* __restrict__ Cre, const float* __restrict__ Cim,
                         unsigned short* __restrict__ W2T) {
  int idx = blockIdx.x * blockDim.x + threadIdx.x;   // over HDIM*N1
  if (idx >= HDIM * N1) return;
  int h = idx / N1;
  int k = idx % N1;
  int p = k >> 1;
  float v = (k & 1) ? (-2.0f * Cim[h * PDIM + p]) : (2.0f * Cre[h * PDIM + p]);
  W2T[(size_t)h * N1 + k] = f32_to_bf16_rne(v);
}

__global__ void prep_lam(const float* __restrict__ Lre, const float* __restrict__ Lim,
                         const float* __restrict__ lstep,
                         float* __restrict__ lamr, float* __restrict__ lami) {
  int p = blockIdx.x * blockDim.x + threadIdx.x;
  if (p >= PDIM) return;
  float dt = __expf(lstep[p]);
  float e  = __expf(Lre[p] * dt);
  lamr[p] = e * __cosf(Lim[p] * dt);
  lami[p] = e * __sinf(Lim[p] * dt);
}

// ---------------- RMSNorm -> u (bf16), vectorized ----------------

__global__ void __launch_bounds__(256) rms_u(const float* __restrict__ hs,
                                             const float* __restrict__ nw,
                                             unsigned short* __restrict__ u16,
                                             float* __restrict__ inv_rms) {
  __shared__ float red[256];
  int row = blockIdx.x;
  int tid = threadIdx.x;
  const float4* x4 = (const float4*)(hs + (size_t)row * HDIM);
  float4 v = x4[tid];                                 // HDIM/4 == 256 == blockDim
  float ss = v.x * v.x + v.y * v.y + v.z * v.z + v.w * v.w;
  red[tid] = ss;
  __syncthreads();
  for (int s = 128; s > 0; s >>= 1) {
    if (tid < s) red[tid] += red[tid + s];
    __syncthreads();
  }
  float inv = rsqrtf(red[0] * (1.0f / HDIM) + EPSV);
  if (tid == 0) inv_rms[row] = inv;
  float4 w = ((const float4*)nw)[tid];
  uint2 o;
  o.x = pack_bf16x2(v.x * inv * w.x, v.y * inv * w.y);
  o.y = pack_bf16x2(v.z * inv * w.z, v.w * inv * w.w);
  ((uint2*)(u16 + (size_t)row * HDIM))[tid] = o;
}

// ---------------- WMMA fragment loads (ISA §7.12.2 layouts) ----------------

// A 16x32 bf16 tile at row stride `stride`, K offset kOff:
// lane l -> row l%16; VGPR v -> K pair (v/4)*16 + (l/16)*8 + (v%4)*2
__device__ __forceinline__ v16bf load_frag_a(const unsigned short* L, int rowBase, int lane,
                                             int stride, int kOff) {
  BF16Frag f;
  const unsigned short* r = L + (size_t)(rowBase + (lane & 15)) * stride + kOff;
  int half = (lane >> 4) & 1;
#pragma unroll
  for (int v = 0; v < 8; ++v) {
    int kp = ((v >> 2) << 4) + (half << 3) + ((v & 3) << 1);
    f.u[v] = *(const unsigned int*)(r + kp);
  }
  return f.v;
}

// B 32x16 bf16 from LDS stored transposed [n][32]: lane l -> col l%16;
// VGPR v holds K pair (v/4)*16 + (l/16)*8 + (v%4)*2
__device__ __forceinline__ v16bf load_frag_b(const unsigned short* L, int colBase, int lane) {
  BF16Frag f;
  const unsigned short* r = L + (colBase + (lane & 15)) * 32;
  int half = (lane >> 4) & 1;
#pragma unroll
  for (int v = 0; v < 8; ++v) {
    int k = ((v >> 2) << 4) + (half << 3) + ((v & 3) << 1);
    f.u[v] = *(const unsigned int*)(r + k);
  }
  return f.v;
}

// ------- GEMM1 (+ fused chunk-carry extraction):
//   Bu16 = bf16( u_bf16 (M x 1024) @ W1T^T (1024 x 512) ), carries per (chunk, p) -------

__global__ void __launch_bounds__(256, 2) gemm_bu(const unsigned short* __restrict__ U,
                                                  const unsigned short* __restrict__ W1T,
                                                  unsigned short* __restrict__ Bu16,
                                                  const float* __restrict__ lamr,
                                                  const float* __restrict__ lami,
                                                  float2* __restrict__ carry) {
  __shared__ alignas(16) unsigned short smem[128 * 128];  // 32KB; K loop uses first 16KB
  unsigned short* lA = smem;               // [m][32]
  unsigned short* lB = smem + 128 * 32;    // [n][32]
  const int tid = threadIdx.x;
  const int lane = tid & 31, wave = tid >> 5;
  const int wm = wave & 1, wn = wave >> 1;               // 2 x 4 wave grid
  const int rowBase = blockIdx.y * 128;                  // == one scan chunk
  const int colBase = blockIdx.x * 128;
  const int mc = tid >> 1;                               // copy row/col (0..127)
  const int kc = (tid & 1) << 4;                         // copy k offset (0 or 16)
  const v8f vzero = {0.f, 0.f, 0.f, 0.f, 0.f, 0.f, 0.f, 0.f};
  v8f acc[4][2];
#pragma unroll
  for (int ms = 0; ms < 4; ++ms)
#pragma unroll
    for (int ns = 0; ns < 2; ++ns) acc[ms][ns] = vzero;

  for (int kt = 0; kt < HDIM / 32; ++kt) {
    cp16b(lA + mc * 32 + kc, U + (size_t)(rowBase + mc) * HDIM + kt * 32 + kc);
    cp16b(lB + mc * 32 + kc, W1T + (size_t)(colBase + mc) * HDIM + kt * 32 + kc);
    if (kt + 1 < HDIM / 32)
      __builtin_prefetch(U + (size_t)(rowBase + mc) * HDIM + (kt + 1) * 32 + kc, 0, 1);
    async_join();
    __syncthreads();
    v16bf bf0 = load_frag_b(lB, wn * 32, lane);
    v16bf bf1 = load_frag_b(lB, wn * 32 + 16, lane);
#pragma unroll
    for (int ms = 0; ms < 4; ++ms) {
      v16bf af = load_frag_a(lA, wm * 64 + ms * 16, lane, 32, 0);
      acc[ms][0] = __builtin_amdgcn_wmma_f32_16x16x32_bf16(false, af, false, bf0,
                                                           (short)0, acc[ms][0], false, false);
      acc[ms][1] = __builtin_amdgcn_wmma_f32_16x16x32_bf16(false, af, false, bf1,
                                                           (short)0, acc[ms][1], false, false);
    }
    __syncthreads();
  }

  // epilogue: accumulators -> bf16 tile in LDS (reuses full 32KB smem)
  const int half = lane >> 4, nlo = lane & 15;
#pragma unroll
  for (int ms = 0; ms < 4; ++ms)
#pragma unroll
    for (int ns = 0; ns < 2; ++ns) {
      int nl = wn * 32 + ns * 16 + nlo;
#pragma unroll
      for (int v = 0; v < 8; ++v) {
        int ml = wm * 64 + ms * 16 + v + 8 * half;
        smem[ml * 128 + nl] = f32_to_bf16_rne(acc[ms][ns][v]);
      }
    }
  __syncthreads();
  // vectorized global write of the bf16 tile
#pragma unroll
  for (int j = 0; j < 8; ++j) {
    int idx = tid + j * 256;                 // 2048 vectors of 8 ushorts
    int m = idx >> 4, v8 = idx & 15;
    *(uint4*)(Bu16 + (size_t)(rowBase + m) * N1 + colBase + v8 * 8) =
        *(const uint4*)(smem + m * 128 + v8 * 8);
  }
  // fused chunk-carry scan: 64 complex columns owned by this block
  if (tid < 64) {
    int q = tid;
    int p = (colBase >> 1) + q;
    float ar = lamr[p], ai = lami[p];
    float xr = 0.f, xi = 0.f;
    for (int i = 0; i < LCHUNK; ++i) {
      unsigned int w = *(const unsigned int*)(smem + i * 128 + 2 * q);
      float nr2 = ar * xr - ai * xi + bf16_lo(w);
      float ni2 = ar * xi + ai * xr + bf16_hi(w);
      xr = nr2; xi = ni2;
    }
    int b = rowBase / SEQL;
    int c = (rowBase % SEQL) / LCHUNK;
    carry[(size_t)(b * NCHUNK + c) * PDIM + p] = make_float2(xr, xi);
  }
}

// ---- sequential scan over chunk carries; also emits final_state ----

__global__ void chunk_scan(const float2* __restrict__ carry,
                           const float* __restrict__ lamr, const float* __restrict__ lami,
                           float2* __restrict__ scanned, float* __restrict__ out_tail) {
  int t = blockIdx.x * blockDim.x + threadIdx.x;   // b*PDIM + p
  if (t >= BATCH * PDIM) return;
  int b = t / PDIM, p = t % PDIM;
  float zr = lamr[p], zi = lami[p];
#pragma unroll
  for (int k = 0; k < 7; ++k) {                    // lam_bar^128
    float nr = zr * zr - zi * zi;
    zi = 2.f * zr * zi;
    zr = nr;
  }
  float xr = 0.f, xi = 0.f;
  for (int c = 0; c < NCHUNK; ++c) {
    size_t idx = (size_t)(b * NCHUNK + c) * PDIM + p;
    float2 cv = carry[idx];
    float nr = zr * xr - zi * xi + cv.x;
    float ni = zr * xi + zi * xr + cv.y;
    xr = nr; xi = ni;
    scanned[idx] = make_float2(xr, xi);
  }
  out_tail[(size_t)t * 2] = xr;                    // final_state (complex64 interleave)
  out_tail[(size_t)t * 2 + 1] = xi;
}

// ------- GEMM2 (+ fused seeded scan): out = hidden + xs @ W2T^T + D*u -------

__global__ void __launch_bounds__(256, 2) gemm_out(const unsigned short* __restrict__ Bu16,
                                                   const unsigned short* __restrict__ W2T,
                                                   const float2* __restrict__ scanned,
                                                   const float* __restrict__ lamr,
                                                   const float* __restrict__ lami,
                                                   const float* __restrict__ hs,
                                                   const float* __restrict__ nw,
                                                   const float* __restrict__ inv_rms,
                                                   const float* __restrict__ Dv,
                                                   float* __restrict__ out) {
  __shared__ alignas(16) unsigned short lAbig[128 * 512];  // full chunk, bf16, 128KB
  __shared__ alignas(16) unsigned short lB[128 * 32];      // 8KB
  const int tid = threadIdx.x;
  const int lane = tid & 31, wave = tid >> 5;
  const int wm = wave & 1, wn = wave >> 1;
  const int rowBase = blockIdx.y * 128;                    // == one scan chunk
  const int colBase = blockIdx.x * 128;
  const int mc = tid >> 1;
  const int kc = (tid & 1) << 4;

  // phase 0: stage the full 128x512 bf16 chunk of Bu into LDS
#pragma unroll
  for (int j = 0; j < 32; ++j) {
    int idx = tid + j * 256;                 // 8192 vectors of 8 ushorts
    cp8b(lAbig + idx * 8, Bu16 + (size_t)rowBase * N1 + idx * 8);
  }
  async_join();
  __syncthreads();

  // phase 1: carry-seeded local scan in LDS; thread p owns complex pair (2p,2p+1)
  {
    int p = tid;
    int b = rowBase / SEQL;
    int c = (rowBase % SEQL) / LCHUNK;
    float ar = lamr[p], ai = lami[p];
    float xr = 0.f, xi = 0.f;
    if (c > 0) {
      float2 s = scanned[(size_t)(b * NCHUNK + c - 1) * PDIM + p];
      xr = s.x; xi = s.y;
    }
    unsigned int* col = (unsigned int*)lAbig + p;          // 256 uints per row
    for (int i = 0; i < LCHUNK; ++i) {
      unsigned int w = col[i * 256];
      float nr2 = ar * xr - ai * xi + bf16_lo(w);
      float ni2 = ar * xi + ai * xr + bf16_hi(w);
      xr = nr2; xi = ni2;
      col[i * 256] = pack_bf16x2(xr, xi);
    }
  }
  __syncthreads();

  // phase 2: K loop; A frags straight from lAbig (stride 512), B staged per kt
  const v8f vzero = {0.f, 0.f, 0.f, 0.f, 0.f, 0.f, 0.f, 0.f};
  v8f acc[4][2];
#pragma unroll
  for (int ms = 0; ms < 4; ++ms)
#pragma unroll
    for (int ns = 0; ns < 2; ++ns) acc[ms][ns] = vzero;

  for (int kt = 0; kt < N1 / 32; ++kt) {
    cp16b(lB + mc * 32 + kc, W2T + (size_t)(colBase + mc) * N1 + kt * 32 + kc);
    async_join();
    __syncthreads();
    v16bf bf0 = load_frag_b(lB, wn * 32, lane);
    v16bf bf1 = load_frag_b(lB, wn * 32 + 16, lane);
#pragma unroll
    for (int ms = 0; ms < 4; ++ms) {
      v16bf af = load_frag_a(lAbig, wm * 64 + ms * 16, lane, N1, kt * 32);
      acc[ms][0] = __builtin_amdgcn_wmma_f32_16x16x32_bf16(false, af, false, bf0,
                                                           (short)0, acc[ms][0], false, false);
      acc[ms][1] = __builtin_amdgcn_wmma_f32_16x16x32_bf16(false, af, false, bf1,
                                                           (short)0, acc[ms][1], false, false);
    }
    __syncthreads();
  }

  const int half = lane >> 4, nlo = lane & 15;
#pragma unroll
  for (int ms = 0; ms < 4; ++ms)
#pragma unroll
    for (int ns = 0; ns < 2; ++ns) {
      int h = colBase + wn * 32 + ns * 16 + nlo;
#pragma unroll
      for (int v = 0; v < 8; ++v) {
        int m = rowBase + wm * 64 + ms * 16 + v + 8 * half;
        size_t gi = (size_t)m * HDIM + h;
        float hv = hs[gi];
        float uu = hv * inv_rms[m] * nw[h];
        out[gi] = hv + Dv[h] * uu + acc[ms][ns][v];
      }
    }
}

// ---------------- launcher ----------------

extern "C" void kernel_launch(void* const* d_in, const int* in_sizes, int n_in,
                              void* d_out, int out_size, void* d_ws, size_t ws_size,
                              hipStream_t stream) {
  const float* hs    = (const float*)d_in[0];
  const float* nw    = (const float*)d_in[1];
  const float* Lre   = (const float*)d_in[2];
  const float* Lim   = (const float*)d_in[3];
  const float* lstep = (const float*)d_in[4];
  const float* Bre   = (const float*)d_in[5];
  const float* Bim   = (const float*)d_in[6];
  const float* Cre   = (const float*)d_in[7];
  const float* Cim   = (const float*)d_in[8];
  const float* Dv    = (const float*)d_in[9];
  float* out = (float*)d_out;

  char* ws = (char*)d_ws;
  size_t off = 0;
  unsigned short* u16  = (unsigned short*)(ws + off); off += (size_t)MROWS * HDIM * 2;
  unsigned short* Bu16 = (unsigned short*)(ws + off); off += (size_t)MROWS * N1 * 2;
  unsigned short* W1T  = (unsigned short*)(ws + off); off += (size_t)N1 * HDIM * 2;
  unsigned short* W2T  = (unsigned short*)(ws + off); off += (size_t)HDIM * N1 * 2;
  float*          inv  = (float*)(ws + off);          off += (size_t)MROWS * 4;
  float*          lamr = (float*)(ws + off);          off += (size_t)PDIM * 4;
  float*          lami = (float*)(ws + off);          off += (size_t)PDIM * 4;
  float2*         car  = (float2*)(ws + off);         off += (size_t)BATCH * NCHUNK * PDIM * 8;
  float2*         scn  = (float2*)(ws + off);         off += (size_t)BATCH * NCHUNK * PDIM * 8;

  prep_w1t<<<(N1 * HDIM + 255) / 256, 256, 0, stream>>>(Lre, Lim, lstep, Bre, Bim, W1T);
  prep_w2t<<<(HDIM * N1 + 255) / 256, 256, 0, stream>>>(Cre, Cim, W2T);
  prep_lam<<<1, 256, 0, stream>>>(Lre, Lim, lstep, lamr, lami);
  rms_u<<<MROWS, 256, 0, stream>>>(hs, nw, u16, inv);
  gemm_bu<<<dim3(N1 / 128, MROWS / 128), 256, 0, stream>>>(u16, W1T, Bu16, lamr, lami, car);
  chunk_scan<<<(BATCH * PDIM + 255) / 256, 256, 0, stream>>>(car, lamr, lami, scn,
                                                             out + (size_t)MROWS * HDIM);
  gemm_out<<<dim3(HDIM / 128, MROWS / 128), 256, 0, stream>>>(Bu16, W2T, scn, lamr, lami,
                                                              hs, nw, inv, Dv, out);
}
